// TransformerBlock_16174846837179
// MI455X (gfx1250) — matrix operand order repaired and checked by reading, not compiled, not run on hardware
//
#include <hip/hip_runtime.h>
#include <hip/hip_fp16.h>
#include <math.h>
#include <stdint.h>

typedef _Float16 h16;
typedef _Float16 v16h __attribute__((ext_vector_type(16)));
typedef _Float16 v8h  __attribute__((ext_vector_type(8)));
typedef _Float16 v4h  __attribute__((ext_vector_type(4)));
typedef float    v8f  __attribute__((ext_vector_type(8)));
typedef float    v4f  __attribute__((ext_vector_type(4)));

#define B_   4
#define T_   1024
#define H_   1024
#define NH_  16
#define DH_  64
#define E_   8
#define S_   (B_ * T_)
#define CAP_ 512
#define H3_  (3 * H_)
#define FF_  (4 * H_)

// ---------------------------------------------------------------------------
// WMMA helpers (CDNA5 gfx1250: V_WMMA_F32_16X16X32_F16, wave32)
// ---------------------------------------------------------------------------
__device__ __forceinline__ v8f wmma_f16(v16h a, v16h b, v8f c) {
  // 8 args: (neg_a, A, neg_b, B, c_mod, C, reuse_a, reuse_b)
  return __builtin_amdgcn_wmma_f32_16x16x32_f16(false, a, false, b, (short)0, c,
                                                false, false);
}

__device__ __forceinline__ v16h mk16(v8h lo, v8h hi) {
  return __builtin_shufflevector(lo, hi, 0, 1, 2, 3, 4, 5, 6, 7, 8, 9, 10, 11,
                                 12, 13, 14, 15);
}

// ---------------------------------------------------------------------------
// CDNA5 async global->LDS copy (ASYNCcnt path). Each lane DMAs 16 bytes from
// global straight into LDS, no VGPR landing. LDS dest = low 32 bits of the
// generic shared-pointer (LDS aperture low bits == LDS address, ISA 10.2).
// ---------------------------------------------------------------------------
__device__ __forceinline__ void async_copy_b128(void* lds_dst, const void* gsrc) {
  uint32_t lds_off = (uint32_t)(uintptr_t)lds_dst;
  asm volatile("global_load_async_to_lds_b128 %0, %1, off"
               :
               : "v"(lds_off), "v"(gsrc)
               : "memory");
}
__device__ __forceinline__ void wait_async0() {
  asm volatile("s_wait_asynccnt 0x0" ::: "memory");
}

// ---------------------------------------------------------------------------
// LayerNorm (fp32 stats) -> f16 output
// ---------------------------------------------------------------------------
__global__ void ln_kernel(const float* __restrict__ x, const float* __restrict__ w,
                          const float* __restrict__ b, h16* __restrict__ out) {
  int s = blockIdx.x;
  const float* row = x + (size_t)s * H_;
  int tid = threadIdx.x;  // 256 threads
  float v[4];
  float sum = 0.f, sq = 0.f;
#pragma unroll
  for (int i = 0; i < 4; i++) {
    v[i] = row[tid + i * 256];
    sum += v[i];
    sq += v[i] * v[i];
  }
#pragma unroll
  for (int off = 1; off < 32; off <<= 1) {
    sum += __shfl_xor(sum, off, 32);
    sq += __shfl_xor(sq, off, 32);
  }
  __shared__ float ssum[8], ssq[8];
  int wave = tid >> 5, lane = tid & 31;
  if (lane == 0) { ssum[wave] = sum; ssq[wave] = sq; }
  __syncthreads();
  sum = 0.f; sq = 0.f;
#pragma unroll
  for (int i = 0; i < 8; i++) { sum += ssum[i]; sq += ssq[i]; }
  float mean = sum * (1.0f / H_);
  float var = sq * (1.0f / H_) - mean * mean;
  float r = rsqrtf(var + 1e-5f);
#pragma unroll
  for (int i = 0; i < 4; i++) {
    int c = tid + i * 256;
    out[(size_t)s * H_ + c] = (h16)((v[i] - mean) * r * w[c] + b[c]);
  }
}

// ---------------------------------------------------------------------------
// Generic tiled WMMA GEMM:  C[M,N] = A[M,K](f16) * B^T + bias
// A tile staged via async global->LDS DMA (overlaps with the f32->f16 B-tile
// conversion which must pass through VGPRs anyway). EPI: 0=f16 out,
// 1=f32 out + residual, 2=f16 GELU, 3=f32 out.
// BM=BN=128, BK=32, 256 threads = 8 waves (2x4).
// ---------------------------------------------------------------------------
template <int EPI>
__global__ void gemm_kernel(const h16* __restrict__ A, const float* __restrict__ Bm,
                            const float* __restrict__ bias,
                            const float* __restrict__ res, void* __restrict__ Out,
                            int M, int N, int K, int ldbn, int ldbk, long aBatch,
                            long bBatch, long biasBatch, long oBatch) {
  __shared__ h16 As[128 * 40];
  __shared__ h16 Bs[128 * 40];
  int e = blockIdx.z;
  const h16* Ab = A + (size_t)e * aBatch;
  const float* Bb = Bm + (size_t)e * bBatch;
  const float* biasb = bias + (size_t)e * biasBatch;
  int m0 = blockIdx.x * 128;
  int n0 = blockIdx.y * 128;
  int tid = threadIdx.x;
  int wave = tid >> 5, lane = tid & 31;
  int wm = wave >> 2, wn = wave & 3;
  int lm = lane & 15;
  int hi = lane >> 4;

  v8f acc[4][2];
#pragma unroll
  for (int i = 0; i < 4; i++)
#pragma unroll
    for (int j = 0; j < 2; j++) acc[i][j] = (v8f){0, 0, 0, 0, 0, 0, 0, 0};

  for (int k0 = 0; k0 < K; k0 += 32) {
    // ---- stage A tile (128x32 f16): fire-and-forget async DMA to LDS ----
    {
      int row = tid >> 2;          // 0..63
      int col = (tid & 3) * 8;     // 0,8,16,24
      const h16* src = Ab + (size_t)(m0 + row) * K + (k0 + col);
      async_copy_b128(&As[row * 40 + col], src);
      async_copy_b128(&As[(row + 64) * 40 + col], src + (size_t)64 * K);
      if (k0 + 32 < K) __builtin_prefetch(src + 32, 0, 1);
    }
    // ---- stage B tile (128 n x 32 k), f32 -> f16 (overlaps async DMA) ----
    if (ldbk == 1) {  // K contiguous (projection weights [N,K])
#pragma unroll
      for (int i = 0; i < 4; i++) {
        int qid = tid + i * 256;   // 0..1023
        int n = qid >> 3;          // 0..127
        int k4 = (qid & 7) * 4;
        v4f f = *(const v4f*)(Bb + (size_t)(n0 + n) * ldbn + (k0 + k4));
        v4h hv;
        hv[0] = (h16)f[0]; hv[1] = (h16)f[1]; hv[2] = (h16)f[2]; hv[3] = (h16)f[3];
        *(v4h*)&Bs[n * 40 + k4] = hv;
      }
    } else {          // N contiguous (expert weights [K,N])
#pragma unroll
      for (int i = 0; i < 4; i++) {
        int qid = tid + i * 256;
        int k = qid >> 5;          // 0..31
        int n4 = (qid & 31) * 4;
        v4f f = *(const v4f*)(Bb + (size_t)(k0 + k) * ldbk + (n0 + n4));
#pragma unroll
        for (int j = 0; j < 4; j++) Bs[(n4 + j) * 40 + k] = (h16)f[j];
      }
    }
    wait_async0();  // A-tile DMA landed in LDS
    __syncthreads();
    // ---- fragments + 8 WMMA ----
    v16h af[4];
#pragma unroll
    for (int i = 0; i < 4; i++) {
      int m = wm * 64 + i * 16 + lm;
      int kk = hi * 8;
      af[i] = mk16(*(const v8h*)&As[m * 40 + kk],
                   *(const v8h*)&As[m * 40 + 16 + kk]);
    }
    v16h bf[2];
#pragma unroll
    for (int j = 0; j < 2; j++) {
      int n = wn * 32 + j * 16 + lm;
      int kk2 = hi * 8;
      bf[j] = mk16(*(const v8h*)&Bs[n * 40 + kk2],
                   *(const v8h*)&Bs[n * 40 + 16 + kk2]);
    }
#pragma unroll
    for (int i = 0; i < 4; i++)
#pragma unroll
      for (int j = 0; j < 2; j++) acc[i][j] = wmma_f16(af[i], bf[j], acc[i][j]);
    __syncthreads();
  }

  // ---- epilogue (C layout: lane n = lm, row = r + 8*hi) ----
#pragma unroll
  for (int i = 0; i < 4; i++) {
    int rowb = m0 + wm * 64 + i * 16 + hi * 8;
#pragma unroll
    for (int j = 0; j < 2; j++) {
      int col = n0 + wn * 32 + j * 16 + lm;
      float bv = biasb[col];
#pragma unroll
      for (int r = 0; r < 8; r++) {
        size_t idx = (size_t)e * oBatch + (size_t)(rowb + r) * N + col;
        float v = acc[i][j][r] + bv;
        if (EPI == 0) {
          ((h16*)Out)[idx] = (h16)v;
        } else if (EPI == 1) {
          ((float*)Out)[idx] = v + res[idx];
        } else if (EPI == 2) {
          v = 0.5f * v * (1.0f + erff(v * 0.70710678118f));
          ((h16*)Out)[idx] = (h16)v;
        } else {
          ((float*)Out)[idx] = v;
        }
      }
    }
  }
}

// ---------------------------------------------------------------------------
// Flash attention: block = 128 threads (4 waves), each wave owns a 16-row q
// tile; block covers 64 q rows of one (b, head). fp32 online softmax.
// ---------------------------------------------------------------------------
__global__ void attn_kernel(const h16* __restrict__ qkv, h16* __restrict__ o) {
  int qb = blockIdx.x, head = blockIdx.y, b = blockIdx.z;
  int tid = threadIdx.x;
  int wave = tid >> 5, lane = tid & 31;
  int lm = lane & 15, hi = lane >> 4;
  int sbase = b * T_;
  int qoff = head * DH_;
  int koff = H_ + head * DH_;
  int voff = 2 * H_ + head * DH_;
  int q0 = qb * 64 + wave * 16;

  __shared__ h16 Vt[64 * 40];        // V^T tile: [dh][k], k-chunk of 32
  __shared__ h16 Pbuf[4 * 16 * 40];  // per-wave P transpose scratch
  h16* Pw = &Pbuf[wave * 16 * 40];

  // Q fragments (16x64) straight from global: two b128 loads per frag
  v16h aq[2];
  {
    const h16* qrow = qkv + (size_t)(sbase + q0 + lm) * H3_ + qoff;
    int kk = hi * 8;
#pragma unroll
    for (int d = 0; d < 2; d++)
      aq[d] = mk16(*(const v8h*)(qrow + d * 32 + kk),
                   *(const v8h*)(qrow + d * 32 + 16 + kk));
  }
  v8f oacc[4];
#pragma unroll
  for (int n = 0; n < 4; n++) oacc[n] = (v8f){0, 0, 0, 0, 0, 0, 0, 0};
  float mrun[8], lrun[8];
#pragma unroll
  for (int r = 0; r < 8; r++) { mrun[r] = -1e30f; lrun[r] = 0.f; }

  for (int kc = 0; kc < T_ / 32; kc++) {
    int k0 = kc * 32;
    // stage V^T (32 k x 64 dh -> Vt[dh][k])
#pragma unroll
    for (int i = 0; i < 16; i++) {
      int idx = tid * 16 + i;   // 0..2047
      int k = idx >> 6;         // fixed per thread
      int dh = idx & 63;
      Vt[dh * 40 + k] = qkv[(size_t)(sbase + k0 + k) * H3_ + voff + dh];
    }
    __syncthreads();

    // scores: two 16x16 tiles over this 32-wide k chunk
    float p[2][8], mx[8];
#pragma unroll
    for (int r = 0; r < 8; r++) mx[r] = -1e30f;
#pragma unroll
    for (int j = 0; j < 2; j++) {
      const h16* krow = qkv + (size_t)(sbase + k0 + j * 16 + lm) * H3_ + koff;
      int kk2 = hi * 8;
      v8f s = (v8f){0, 0, 0, 0, 0, 0, 0, 0};
#pragma unroll
      for (int d = 0; d < 2; d++) {
        v16h bk = mk16(*(const v8h*)(krow + d * 32 + kk2),
                       *(const v8h*)(krow + d * 32 + 16 + kk2));
        s = wmma_f16(aq[d], bk, s);
      }
#pragma unroll
      for (int r = 0; r < 8; r++) {
        float v = s[r] * 0.125f;  // 1/sqrt(64)
        p[j][r] = v;
        mx[r] = fmaxf(mx[r], v);
      }
    }
    // row reductions within 16-lane groups (rows r / r+8 per half-wave)
#pragma unroll
    for (int r = 0; r < 8; r++) {
      float v = mx[r];
      v = fmaxf(v, __shfl_xor(v, 1, 16));
      v = fmaxf(v, __shfl_xor(v, 2, 16));
      v = fmaxf(v, __shfl_xor(v, 4, 16));
      v = fmaxf(v, __shfl_xor(v, 8, 16));
      float mn = fmaxf(mrun[r], v);
      float alpha = __expf(mrun[r] - mn);
      mrun[r] = mn;
      float s0 = __expf(p[0][r] - mn);
      float s1 = __expf(p[1][r] - mn);
      p[0][r] = s0; p[1][r] = s1;
      float t = s0 + s1;
      t += __shfl_xor(t, 1, 16);
      t += __shfl_xor(t, 2, 16);
      t += __shfl_xor(t, 4, 16);
      t += __shfl_xor(t, 8, 16);
      lrun[r] = alpha * lrun[r] + t;
#pragma unroll
      for (int n = 0; n < 4; n++) oacc[n][r] *= alpha;
    }
    // P (C-layout) -> f16 A-layout via per-wave LDS scratch
#pragma unroll
    for (int j = 0; j < 2; j++)
#pragma unroll
      for (int r = 0; r < 8; r++)
        Pw[(r + hi * 8) * 40 + j * 16 + lm] = (h16)p[j][r];
    __builtin_amdgcn_wave_barrier();  // keep ds stores before ds loads
    v16h ap;
    {
      int kk = hi * 8;
      ap = mk16(*(const v8h*)&Pw[lm * 40 + kk],
                *(const v8h*)&Pw[lm * 40 + 16 + kk]);
    }
    // O += P * V  (4 dh tiles of 16)
#pragma unroll
    for (int n = 0; n < 4; n++) {
      int kk2 = hi * 8;
      v16h bv = mk16(*(const v8h*)&Vt[(n * 16 + lm) * 40 + kk2],
                     *(const v8h*)&Vt[(n * 16 + lm) * 40 + 16 + kk2]);
      oacc[n] = wmma_f16(ap, bv, oacc[n]);
    }
    __syncthreads();
  }
  // epilogue: O /= l, write f16 [S, H]
#pragma unroll
  for (int r = 0; r < 8; r++) {
    float inv = 1.0f / lrun[r];
    int row = q0 + r + hi * 8;
#pragma unroll
    for (int n = 0; n < 4; n++)
      o[(size_t)(sbase + row) * H_ + qoff + n * 16 + lm] =
          (h16)(oacc[n][r] * inv);
  }
}

// ---------------------------------------------------------------------------
// MoE gating: one wave per token; softmax over E=8, argmax (first-max wins)
// ---------------------------------------------------------------------------
__global__ void gate_kernel(const h16* __restrict__ h2, const float* __restrict__ wg,
                            int* __restrict__ eidx, float* __restrict__ gval) {
  int s = blockIdx.x * 8 + (threadIdx.x >> 5);
  int lane = threadIdx.x & 31;
  const h16* row = h2 + (size_t)s * H_;
  float acc[8] = {0, 0, 0, 0, 0, 0, 0, 0};
  for (int i = 0; i < 32; i++) {
    int k = i * 32 + lane;
    float xv = (float)row[k];
    const float* wr = wg + (size_t)k * E_;
    v4f w0 = *(const v4f*)wr;
    v4f w1 = *(const v4f*)(wr + 4);
#pragma unroll
    for (int e = 0; e < 4; e++) acc[e] += xv * w0[e];
#pragma unroll
    for (int e = 0; e < 4; e++) acc[4 + e] += xv * w1[e];
  }
#pragma unroll
  for (int e = 0; e < 8; e++)
#pragma unroll
    for (int off = 1; off < 32; off <<= 1) acc[e] += __shfl_xor(acc[e], off, 32);
  if (lane == 0) {
    float mx = acc[0];
    int am = 0;
#pragma unroll
    for (int e = 1; e < 8; e++)
      if (acc[e] > mx) { mx = acc[e]; am = e; }
    float den = 0.f;
#pragma unroll
    for (int e = 0; e < 8; e++) den += __expf(acc[e] - mx);
    gval[s] = 1.0f / den;  // softmax value at argmax
    eidx[s] = am;
  }
}

__global__ void init_slots_kernel(int* __restrict__ slot_token) {
  slot_token[blockIdx.x * 256 + threadIdx.x] = -1;
}

// Deterministic first-come slot assignment: wave e scans tokens with ballot
__global__ void route_kernel(const int* __restrict__ eidx, float* __restrict__ gval,
                             int* __restrict__ slot_token,
                             int* __restrict__ token_slot) {
  int e = threadIdx.x >> 5;   // 8 waves, one per expert
  int lane = threadIdx.x & 31;
  int cnt = 0;
  for (int base = 0; base < S_; base += 32) {
    int s = base + lane;
    bool match = (eidx[s] == e);
    unsigned int m32 = (unsigned int)__ballot(match);
    int prefix = __popc(m32 & ((1u << lane) - 1u));
    if (match) {
      int pos = cnt + prefix;
      int posc = pos < CAP_ ? pos : CAP_ - 1;
      token_slot[s] = e * CAP_ + posc;
      if (pos < CAP_)
        slot_token[e * CAP_ + pos] = s;
      else
        gval[s] = 0.0f;  // capacity drop
    }
    cnt += __popc(m32);
  }
}

__global__ void dispatch_kernel(const int* __restrict__ slot_token,
                                const h16* __restrict__ h2, h16* __restrict__ disp) {
  int slot = blockIdx.x;
  int s = slot_token[slot];
  int tid = threadIdx.x;  // 128
#pragma unroll
  for (int i = 0; i < 8; i++) {
    int c = tid + i * 128;
    disp[(size_t)slot * H_ + c] = (s >= 0) ? h2[(size_t)s * H_ + c] : (h16)0.0f;
  }
}

__global__ void combine_kernel(const float* __restrict__ x1,
                               const float* __restrict__ eout,
                               const int* __restrict__ token_slot,
                               const float* __restrict__ gval,
                               float* __restrict__ y) {
  int s = blockIdx.x;
  int tid = threadIdx.x;  // 256
  int slot = token_slot[s];
  float g = gval[s];
#pragma unroll
  for (int i = 0; i < 4; i++) {
    int c = tid + i * 256;
    size_t idx = (size_t)s * H_ + c;
    y[idx] = x1[idx] + g * eout[(size_t)slot * H_ + c];
  }
}

// ---------------------------------------------------------------------------
extern "C" void kernel_launch(void* const* d_in, const int* in_sizes, int n_in,
                              void* d_out, int out_size, void* d_ws,
                              size_t ws_size, hipStream_t stream) {
  (void)in_sizes; (void)n_in; (void)out_size; (void)ws_size;
  const float* x     = (const float*)d_in[0];
  const float* ln1_w = (const float*)d_in[1];
  const float* ln1_b = (const float*)d_in[2];
  const float* in_w  = (const float*)d_in[3];
  const float* in_b  = (const float*)d_in[4];
  const float* out_w = (const float*)d_in[5];
  const float* out_b = (const float*)d_in[6];
  const float* ln2_w = (const float*)d_in[7];
  const float* ln2_b = (const float*)d_in[8];
  const float* wg    = (const float*)d_in[9];
  const float* w1    = (const float*)d_in[10];
  const float* b1    = (const float*)d_in[11];
  const float* w2    = (const float*)d_in[12];
  const float* b2    = (const float*)d_in[13];
  float* y = (float*)d_out;

  char* p = (char*)d_ws;
  auto alloc = [&](size_t bytes) -> char* {
    char* r = p;
    p += (bytes + 255) & ~(size_t)255;
    return r;
  };
  h16*   h1         = (h16*)alloc((size_t)S_ * H_ * 2);
  h16*   qkv        = (h16*)alloc((size_t)S_ * H3_ * 2);
  h16*   oatt       = (h16*)alloc((size_t)S_ * H_ * 2);
  float* x1         = (float*)alloc((size_t)S_ * H_ * 4);
  h16*   h2         = (h16*)alloc((size_t)S_ * H_ * 2);
  int*   eidx       = (int*)alloc((size_t)S_ * 4);
  float* gval       = (float*)alloc((size_t)S_ * 4);
  int*   slot_token = (int*)alloc((size_t)E_ * CAP_ * 4);
  int*   token_slot = (int*)alloc((size_t)S_ * 4);
  h16*   disp       = (h16*)alloc((size_t)E_ * CAP_ * H_ * 2);
  h16*   hmid       = (h16*)alloc((size_t)E_ * CAP_ * FF_ * 2);
  float* eout       = (float*)alloc((size_t)E_ * CAP_ * H_ * 4);

  // 1. LN1
  ln_kernel<<<S_, 256, 0, stream>>>(x, ln1_w, ln1_b, h1);
  // 2. QKV projection: [S,H] x [3H,H]^T -> f16 [S,3H]
  gemm_kernel<0><<<dim3(S_ / 128, H3_ / 128, 1), 256, 0, stream>>>(
      h1, in_w, in_b, nullptr, qkv, S_, H3_, H_, H_, 1, 0, 0, 0, 0);
  // 3. attention
  attn_kernel<<<dim3(T_ / 64, NH_, B_), 128, 0, stream>>>(qkv, oatt);
  // 4. out projection + residual -> x1 (f32)
  gemm_kernel<1><<<dim3(S_ / 128, H_ / 128, 1), 256, 0, stream>>>(
      oatt, out_w, out_b, x, x1, S_, H_, H_, H_, 1, 0, 0, 0, 0);
  // 5. LN2
  ln_kernel<<<S_, 256, 0, stream>>>(x1, ln2_w, ln2_b, h2);
  // 6. gating
  gate_kernel<<<S_ / 8, 256, 0, stream>>>(h2, wg, eidx, gval);
  // 7. routing
  init_slots_kernel<<<(E_ * CAP_) / 256, 256, 0, stream>>>(slot_token);
  route_kernel<<<1, 256, 0, stream>>>(eidx, gval, slot_token, token_slot);
  // 8. dispatch
  dispatch_kernel<<<E_ * CAP_, 128, 0, stream>>>(slot_token, h2, disp);
  // 9. expert FFN up + GELU: per expert [CAP,H] x [H,4H] -> f16
  gemm_kernel<2><<<dim3(CAP_ / 128, FF_ / 128, E_), 256, 0, stream>>>(
      disp, w1, b1, nullptr, hmid, CAP_, FF_, H_, 1, FF_, (long)CAP_ * H_,
      (long)H_ * FF_, (long)FF_, (long)CAP_ * FF_);
  // 10. expert FFN down: per expert [CAP,4H] x [4H,H] -> f32
  gemm_kernel<3><<<dim3(CAP_ / 128, H_ / 128, E_), 256, 0, stream>>>(
      hmid, w2, b2, nullptr, eout, CAP_, H_, FF_, 1, H_, (long)CAP_ * FF_,
      (long)FF_ * H_, (long)H_, (long)CAP_ * H_);
  // 11. combine + residual -> output
  combine_kernel<<<S_, 256, 0, stream>>>(x1, eout, token_slot, gval, y);
}
